// MultiHeadedAttention_47931835023376
// MI455X (gfx1250) — matrix operand order repaired and checked by reading, not compiled, not run on hardware
//
#include <hip/hip_runtime.h>
#include <hip/hip_bf16.h>

#define D_MODEL 1024
#define HID     64
#define SEQ     4096
#define BATCH   8
#define NCHUNK  32           // softmax partial chunks per batch (128 rows each)

typedef __attribute__((ext_vector_type(16))) __bf16 v16bf;
typedef __attribute__((ext_vector_type(8)))  __bf16 v8bf;
typedef __attribute__((ext_vector_type(8)))  float  v8f;
typedef __attribute__((ext_vector_type(4)))  float  f4;

// ---- helpers -------------------------------------------------------------

static __device__ __forceinline__ v8bf cvt8(f4 a, f4 b) {
    v8bf r;
    r[0] = (__bf16)a[0]; r[1] = (__bf16)a[1];
    r[2] = (__bf16)a[2]; r[3] = (__bf16)a[3];
    r[4] = (__bf16)b[0]; r[5] = (__bf16)b[1];
    r[6] = (__bf16)b[2]; r[7] = (__bf16)b[3];
    return r;
}

static __device__ __forceinline__ v16bf cat16(v8bf lo, v8bf hi) {
    return __builtin_shufflevector(lo, hi, 0,1,2,3,4,5,6,7,8,9,10,11,12,13,14,15);
}

// A fragment: 16x32 bf16 tile of row-major f32 matrix X (row stride ld).
// ISA layout (16-bit A 16x32): lanes 0-15 -> M=lane,    K = kb + {0..7,16..23};
//                              lanes 16-31 -> M=lane-16, K = kb + {8..15,24..31}.
static __device__ __forceinline__ v16bf load_fragA(const float* __restrict__ X,
                                                   int ld, int kb, int lane) {
    int r  = lane & 15;
    int hs = lane >> 4;
    const float* p = X + (size_t)r * ld + kb + hs * 8;
    f4 a0 = *(const f4*)(p);
    f4 a1 = *(const f4*)(p + 4);
    f4 a2 = *(const f4*)(p + 16);
    f4 a3 = *(const f4*)(p + 20);
    return cat16(cvt8(a0, a1), cvt8(a2, a3));
}

// ---- weight pre-pack: f32 row-major -> bf16 fragment-major ---------------
// B fragment for (nTile, kstep): lane l holds n = nTile*16 + (l&15),
// K = kstep*32 + (l>>4)*8 + {0..7,16..23}.  Packed at ((nTile*ksteps+kstep)*32+l)*16.
__global__ __launch_bounds__(256) void pack_kernel(const float* __restrict__ W,
                                                   int K, int ksteps,
                                                   __bf16* __restrict__ out) {
    int tid   = blockIdx.x * blockDim.x + threadIdx.x;   // one thread per (frag,lane)
    int lane  = tid & 31;
    int frag  = tid >> 5;
    int kstep = frag % ksteps;
    int nTile = frag / ksteps;
    int row   = nTile * 16 + (lane & 15);
    int col   = kstep * 32 + (lane >> 4) * 8;
    const float* p = W + (size_t)row * K + col;
    f4 a0 = *(const f4*)(p);
    f4 a1 = *(const f4*)(p + 4);
    f4 a2 = *(const f4*)(p + 16);
    f4 a3 = *(const f4*)(p + 20);
    v8bf* o = (v8bf*)(out + (size_t)tid * 16);
    o[0] = cvt8(a0, a1);
    o[1] = cvt8(a2, a3);
}

// ---- kernel 1: X[b,N,D] * W^T + bias -> out[b,N,64] ----------------------
// Wave owns 16 rows x 64 cols (4 accumulators); workgroup = 8 waves = 128 rows.
__global__ __launch_bounds__(256) void proj_kernel(const float* __restrict__ X,
                                                   const __bf16* __restrict__ Wpk,
                                                   const float* __restrict__ bias,
                                                   float* __restrict__ out) {
    int b     = blockIdx.y;
    int tid   = threadIdx.x;
    int lane  = tid & 31;
    int w     = tid >> 5;
    int mBase = blockIdx.x * 128 + w * 16;
    const float*  Xb = X + ((size_t)b * SEQ + mBase) * D_MODEL;
    const __bf16* wp = Wpk + (size_t)lane * 16;   // + (nTile*32+kstep)*512 elems

    v8f c0 = {}, c1 = {}, c2 = {}, c3 = {};
    for (int ks = 0; ks < 32; ++ks) {
        v16bf a  = load_fragA(Xb, D_MODEL, ks * 32, lane);
        v16bf b0 = *(const v16bf*)(wp + (size_t)(0 * 32 + ks) * 512);
        v16bf b1 = *(const v16bf*)(wp + (size_t)(1 * 32 + ks) * 512);
        v16bf b2 = *(const v16bf*)(wp + (size_t)(2 * 32 + ks) * 512);
        v16bf b3 = *(const v16bf*)(wp + (size_t)(3 * 32 + ks) * 512);
        c0 = __builtin_amdgcn_wmma_f32_16x16x32_bf16(false, a, false, b0, (short)0, c0, false, false);
        c1 = __builtin_amdgcn_wmma_f32_16x16x32_bf16(false, a, false, b1, (short)0, c1, false, false);
        c2 = __builtin_amdgcn_wmma_f32_16x16x32_bf16(false, a, false, b2, (short)0, c2, false, false);
        c3 = __builtin_amdgcn_wmma_f32_16x16x32_bf16(false, a, false, b3, (short)0, c3, false, false);
    }

    int n  = lane & 15;
    int hs = lane >> 4;
    float* o = out + ((size_t)b * SEQ + mBase + hs * 8) * HID + n;
    v8f cc[4] = {c0, c1, c2, c3};
#pragma unroll
    for (int t = 0; t < 4; ++t) {
        float bv = bias[t * 16 + n];
#pragma unroll
        for (int r = 0; r < 8; ++r)
            o[(size_t)r * HID + t * 16] = cc[t][r] + bv;
    }
}

// ---- kernel 2a: partial softmax-pool over 128-row chunks -----------------
// grid (NCHUNK, BATCH); block reduces 128 rows -> (max, sum_exp, sum_exp*v)[64]
__global__ __launch_bounds__(256) void pool_partial(const float* __restrict__ kp,
                                                    const float* __restrict__ vp,
                                                    float* __restrict__ pmax,
                                                    float* __restrict__ pse,
                                                    float* __restrict__ psv) {
    int chunk = blockIdx.x;
    int b     = blockIdx.y;
    int tid   = threadIdx.x;
    int h     = tid & 63;
    int seg   = tid >> 6;                 // 4 segments x 32 rows
    __shared__ float rmax[4][64];
    __shared__ float rse[4][64];
    __shared__ float rsv[4][64];

    const size_t base = ((size_t)b * SEQ + (size_t)chunk * 128 + (size_t)seg * 32) * HID + h;
    const float* kpp = kp + base;
    const float* vpp = vp + base;

    float m = -3.4e38f;
#pragma unroll 4
    for (int nn = 0; nn < 32; ++nn)
        m = fmaxf(m, kpp[(size_t)nn * HID]);
    rmax[seg][h] = m;
    __syncthreads();
    float cm = fmaxf(fmaxf(rmax[0][h], rmax[1][h]), fmaxf(rmax[2][h], rmax[3][h]));

    float se = 0.f, sv = 0.f;
#pragma unroll 4
    for (int nn = 0; nn < 32; ++nn) {
        float e = __expf(kpp[(size_t)nn * HID] - cm);
        se += e;
        sv += e * vpp[(size_t)nn * HID];
    }
    rse[seg][h] = se;
    rsv[seg][h] = sv;
    __syncthreads();
    if (seg == 0) {
        size_t o = ((size_t)b * NCHUNK + chunk) * HID + h;
        pmax[o] = cm;
        pse[o]  = rse[0][h] + rse[1][h] + rse[2][h] + rse[3][h];
        psv[o]  = rsv[0][h] + rsv[1][h] + rsv[2][h] + rsv[3][h];
    }
}

// ---- kernel 2b: combine chunk partials -> weights[B,64] ------------------
__global__ __launch_bounds__(64) void pool_final(const float* __restrict__ pmax,
                                                 const float* __restrict__ pse,
                                                 const float* __restrict__ psv,
                                                 float* __restrict__ wts) {
    int b = blockIdx.x;
    int h = threadIdx.x;
    const float* pm = pmax + (size_t)b * NCHUNK * HID + h;
    const float* ps = pse  + (size_t)b * NCHUNK * HID + h;
    const float* pv = psv  + (size_t)b * NCHUNK * HID + h;

    float M = -3.4e38f;
#pragma unroll 8
    for (int c = 0; c < NCHUNK; ++c)
        M = fmaxf(M, pm[(size_t)c * HID]);
    float se = 0.f, sv = 0.f;
#pragma unroll 8
    for (int c = 0; c < NCHUNK; ++c) {
        float s = __expf(pm[(size_t)c * HID] - M);
        se += s * ps[(size_t)c * HID];
        sv += s * pv[(size_t)c * HID];
    }
    wts[b * HID + h] = sv / se;
}

// ---- kernel 3: q-proj -> sigmoid gate x weights -> out-proj (fused) ------
// Wave owns 16 rows end-to-end; private LDS tile does the C->A transpose.
__global__ __launch_bounds__(256) void out_kernel(const float* __restrict__ Q,
                                                  const __bf16* __restrict__ Wqpk,
                                                  const float* __restrict__ bq,
                                                  const float* __restrict__ wts,
                                                  const __bf16* __restrict__ Wppk,
                                                  const float* __restrict__ bp,
                                                  float* __restrict__ out) {
    __shared__ __align__(16) __bf16 gld[8][16][72];   // per-wave 16x64 (+8 pad) bf16

    int b     = blockIdx.y;
    int tid   = threadIdx.x;
    int lane  = tid & 31;
    int w     = tid >> 5;
    int mBase = blockIdx.x * 128 + w * 16;
    int n     = lane & 15;
    int hs    = lane >> 4;

    // ---- phase 1: q = Q*Wq^T + bq ; g = sigmoid(q)*weights -> LDS ----
    {
        const float*  Qb = Q + ((size_t)b * SEQ + mBase) * D_MODEL;
        const __bf16* wp = Wqpk + (size_t)lane * 16;
        v8f c0 = {}, c1 = {}, c2 = {}, c3 = {};
        for (int ks = 0; ks < 32; ++ks) {
            v16bf a  = load_fragA(Qb, D_MODEL, ks * 32, lane);
            v16bf b0 = *(const v16bf*)(wp + (size_t)(0 * 32 + ks) * 512);
            v16bf b1 = *(const v16bf*)(wp + (size_t)(1 * 32 + ks) * 512);
            v16bf b2 = *(const v16bf*)(wp + (size_t)(2 * 32 + ks) * 512);
            v16bf b3 = *(const v16bf*)(wp + (size_t)(3 * 32 + ks) * 512);
            c0 = __builtin_amdgcn_wmma_f32_16x16x32_bf16(false, a, false, b0, (short)0, c0, false, false);
            c1 = __builtin_amdgcn_wmma_f32_16x16x32_bf16(false, a, false, b1, (short)0, c1, false, false);
            c2 = __builtin_amdgcn_wmma_f32_16x16x32_bf16(false, a, false, b2, (short)0, c2, false, false);
            c3 = __builtin_amdgcn_wmma_f32_16x16x32_bf16(false, a, false, b3, (short)0, c3, false, false);
        }
        v8f cc[4] = {c0, c1, c2, c3};
#pragma unroll
        for (int t = 0; t < 4; ++t) {
            int   col = t * 16 + n;
            float bqv = bq[col];
            float wv  = wts[b * HID + col];
#pragma unroll
            for (int r = 0; r < 8; ++r) {
                float qv = cc[t][r] + bqv;
                float g  = wv / (1.f + __expf(-qv));   // sigmoid(q) * weights
                gld[w][hs * 8 + r][col] = (__bf16)g;
            }
        }
    }
    __syncthreads();

    // ---- phase 2: out = g * Wp^T + bp ----
    {
        // A fragments from this wave's private LDS tile (K = 64: two k-steps)
        v16bf A0 = cat16(*(const v8bf*)&gld[w][n][hs * 8],
                         *(const v8bf*)&gld[w][n][16 + hs * 8]);
        v16bf A1 = cat16(*(const v8bf*)&gld[w][n][32 + hs * 8],
                         *(const v8bf*)&gld[w][n][48 + hs * 8]);

        const __bf16* wp = Wppk + (size_t)lane * 16;  // + (dTile*2+kstep)*512
        size_t orow = (size_t)b * SEQ + mBase + hs * 8;
        for (int t = 0; t < 64; ++t) {
            v16bf B0 = *(const v16bf*)(wp + (size_t)(t * 2 + 0) * 512);
            v16bf B1 = *(const v16bf*)(wp + (size_t)(t * 2 + 1) * 512);
            v8f c = {};
            c = __builtin_amdgcn_wmma_f32_16x16x32_bf16(false, A0, false, B0, (short)0, c, false, false);
            c = __builtin_amdgcn_wmma_f32_16x16x32_bf16(false, A1, false, B1, (short)0, c, false, false);
            int   dcol = t * 16 + n;
            float bpv  = bp[dcol];
            float* o = out + orow * D_MODEL + dcol;
#pragma unroll
            for (int rr = 0; rr < 8; ++rr)
                o[(size_t)rr * D_MODEL] = c[rr] + bpv;
        }
    }
}

// ---- host launch ---------------------------------------------------------

extern "C" void kernel_launch(void* const* d_in, const int* in_sizes, int n_in,
                              void* d_out, int out_size, void* d_ws, size_t ws_size,
                              hipStream_t stream) {
    const float* query = (const float*)d_in[0];
    const float* key   = (const float*)d_in[1];
    const float* value = (const float*)d_in[2];
    const float* Wq    = (const float*)d_in[3];
    const float* bq    = (const float*)d_in[4];
    const float* Wk    = (const float*)d_in[5];
    const float* bk    = (const float*)d_in[6];
    const float* Wv    = (const float*)d_in[7];
    const float* bv    = (const float*)d_in[8];
    const float* Wp    = (const float*)d_in[9];
    const float* bp    = (const float*)d_in[10];
    float* out = (float*)d_out;

    const size_t PROJ_ELEMS = (size_t)BATCH * SEQ * HID;   // 2M floats
    const size_t PART_ELEMS = (size_t)BATCH * NCHUNK * HID; // 16K floats
    float* kproj   = (float*)d_ws;
    float* vproj   = kproj + PROJ_ELEMS;
    float* weights = vproj + PROJ_ELEMS;                   // 512 floats
    float* pmax    = weights + 512;
    float* pse     = pmax + PART_ELEMS;
    float* psv     = pse + PART_ELEMS;
    __bf16* pWq = (__bf16*)(psv + PART_ELEMS);             // 64K bf16 each, 32B-aligned
    __bf16* pWk = pWq + 65536;
    __bf16* pWv = pWk + 65536;
    __bf16* pWp = pWv + 65536;

    // pre-pack weights to fragment-major bf16 (4096 threads each)
    pack_kernel<<<16, 256, 0, stream>>>(Wq, D_MODEL, 32, pWq);
    pack_kernel<<<16, 256, 0, stream>>>(Wk, D_MODEL, 32, pWk);
    pack_kernel<<<16, 256, 0, stream>>>(Wv, D_MODEL, 32, pWv);
    pack_kernel<<<16, 256, 0, stream>>>(Wp, HID,     2,  pWp);

    dim3 grid(SEQ / 128, BATCH);
    proj_kernel<<<grid, 256, 0, stream>>>(key,   pWk, bk, kproj);
    proj_kernel<<<grid, 256, 0, stream>>>(value, pWv, bv, vproj);

    dim3 pgrid(NCHUNK, BATCH);
    pool_partial<<<pgrid, 256, 0, stream>>>(kproj, vproj, pmax, pse, psv);
    pool_final<<<BATCH, 64, 0, stream>>>(pmax, pse, psv, weights);

    out_kernel<<<grid, 256, 0, stream>>>(query, pWq, bq, weights, pWp, bp, out);
}
